// DiffusionDetHead_3642132267626
// MI455X (gfx1250) — matrix operand order repaired and checked by reading, not compiled, not run on hardware
//
#include <hip/hip_runtime.h>
#include <math.h>

// ---------------- constants ----------------
#define NIMG 2
#define NRP  300
#define MTOT 600          // NIMG*NRP
#define DMOD 256
#define NHEAD 8
#define HD   32
#define DFF_ 2048
#define RR   49           // 7*7
#define DDYN 64
#define NPAR 32768        // 2*DMOD*DDYN
#define SCALE_CLAMP_F 4.135166556742356f

typedef __attribute__((ext_vector_type(16))) _Float16 v16h;
typedef __attribute__((ext_vector_type(8)))  _Float16 v8h;
typedef __attribute__((ext_vector_type(4)))  _Float16 v4h;
typedef __attribute__((ext_vector_type(8)))  float    v8f;

// =====================================================================
// Generic batched WMMA GEMM on f16 operands, f32 accumulate:
//   C[b] = op(A[b]) * B[b] (+bias) (+relu),  A,B are _Float16
//   A element (r,c):  AT ? A[c*lda + r] : A[r*lda + c]      (A is M x K)
//   B element (k,n):  BT ? B[n*ldb + k] : B[k*ldb + n]      (B is K x N)
// One wave computes a 16x64 C tile = 4 x v_wmma_f32_16x16x32_f16 per
// K-step, reusing the A fragment. All 4 B fragments are loaded before
// the WMMAs (10 clustered b128 loads / K-step, zero cvt). OOB rows/cols
// handled by address clamping; K must be a multiple of 32.
// Output: f32 to C (if non-null) and/or f16 to Ch (if non-null); the
// null-checks and relu are wave-uniform and hoisted out of the store
// loops so the epilogue is branch-light.
// =====================================================================
template<int AT, int BT>
__global__ __launch_bounds__(256) void gemm_wmma_kernel(
    const _Float16* __restrict__ A, long long sA, int lda,
    const _Float16* __restrict__ B, long long sB, int ldb,
    const float* __restrict__ bias,
    float* __restrict__ C, _Float16* __restrict__ Ch, long long sC, int ldc,
    int M, int N, int K, int relu)
{
  const int wid  = threadIdx.x >> 5;
  const int lane = threadIdx.x & 31;
  const int nTn = (N + 63) >> 6;          // 64-wide column tiles
  const int nTm = (M + 15) >> 4;
  const int tile = blockIdx.x * 8 + wid;
  if (tile >= nTm * nTn) return;          // wave-uniform
  const int mt = tile / nTn, nt = tile % nTn;
  const long long bat = blockIdx.y;
  A += bat * sA; B += bat * sB;

  const int m0 = mt << 4, n0 = nt << 6;
  const int lhalf = lane >> 4;            // 0: lanes 0-15, 1: lanes 16-31
  const int l15   = lane & 15;
  const int aRow  = min(m0 + l15, M - 1); // clamped: safe unconditional loads

  int nCol[4], nColc[4];
#pragma unroll
  for (int s = 0; s < 4; ++s) {
    nCol[s]  = n0 + 16 * s + l15;
    nColc[s] = min(nCol[s], N - 1);
  }

  v8f acc[4];
#pragma unroll
  for (int s = 0; s < 4; ++s) {
    const float bv = bias ? bias[nColc[s]] : 0.0f;
#pragma unroll
    for (int r = 0; r < 8; ++r) acc[s][r] = bv;
  }

  for (int kk = 0; kk < K; kk += 32) {
    // ---- A fragment: 16-bit A 16x32 layout (ISA 7.12.2) ----
    v16h af;
    if (AT == 0) {
      const _Float16* Ap = A + (long long)aRow * lda + kk + (lhalf << 3);
      const v8h lo = *(const v8h*)Ap;          // K = kk+8h .. +7
      const v8h hi = *(const v8h*)(Ap + 16);   // K = kk+16+8h .. +7
      af = __builtin_shufflevector(lo, hi, 0, 1, 2, 3, 4, 5, 6, 7,
                                   8, 9, 10, 11, 12, 13, 14, 15);
    } else {
      const _Float16* Ap = A + aRow;
#pragma unroll
      for (int i = 0; i < 8; ++i) {
        const int k0 = kk + ((i >> 2) << 4) + (lhalf << 3) + ((i & 3) << 1);
        af[2 * i]     = Ap[(long long)k0 * lda];
        af[2 * i + 1] = Ap[(long long)(k0 + 1) * lda];
      }
    }
    // ---- load ALL 4 B fragments first (same K order as A: kk+8h..+7, kk+16+8h..+7) ----
    const int kb = kk + (lhalf << 3);
    v16h bf[4];
#pragma unroll
    for (int s = 0; s < 4; ++s) {
      if (BT == 1) {
        const _Float16* Bp = B + (long long)nColc[s] * ldb + kb;
        const v8h lo = *(const v8h*)Bp;
        const v8h hi = *(const v8h*)(Bp + 16);
        bf[s] = __builtin_shufflevector(lo, hi, 0, 1, 2, 3, 4, 5, 6, 7,
                                        8, 9, 10, 11, 12, 13, 14, 15);
      } else {
        const _Float16* Bp = B + (long long)kb * ldb + nColc[s];
#pragma unroll
        for (int j = 0; j < 16; ++j) bf[s][j] = Bp[(long long)((j & 7) + ((j >> 3) << 4)) * ldb];
      }
    }
    // ---- 4 WMMAs sharing the A fragment ----
#pragma unroll
    for (int s = 0; s < 4; ++s)
      acc[s] = __builtin_amdgcn_wmma_f32_16x16x32_f16(
          false, af, false, bf[s], (short)0, acc[s], false, false);
  }

  // ---- epilogue: uniform relu/null checks hoisted out of store loops ----
  if (relu) {
#pragma unroll
    for (int s = 0; s < 4; ++s)
#pragma unroll
      for (int r = 0; r < 8; ++r) acc[s][r] = fmaxf(acc[s][r], 0.0f);
  }
  const int rowBase = m0 + (lhalf << 3);  // VGPR r -> row rowBase + r
  if (C) {
    float* Cb = C + bat * sC;
#pragma unroll
    for (int s = 0; s < 4; ++s) {
      if (nCol[s] < N) {
#pragma unroll
        for (int r = 0; r < 8; ++r) {
          if (rowBase + r < M)
            Cb[(long long)(rowBase + r) * ldc + nCol[s]] = acc[s][r];
        }
      }
    }
  }
  if (Ch) {
    _Float16* Cb = Ch + bat * sC;
#pragma unroll
    for (int s = 0; s < 4; ++s) {
      if (nCol[s] < N) {
#pragma unroll
        for (int r = 0; r < 8; ++r) {
          if (rowBase + r < M)
            Cb[(long long)(rowBase + r) * ldc + nCol[s]] = (_Float16)acc[s][r];
        }
      }
    }
  }
}

// f32 -> f16 bulk convert (4 elements/thread)
__global__ __launch_bounds__(256) void cvt_f16_kernel(
    const float* __restrict__ x, _Float16* __restrict__ y, int n4)
{
  const int i = blockIdx.x * 256 + threadIdx.x;
  if (i < n4) {
    const float4 v = ((const float4*)x)[i];
    v4h h;
    h[0] = (_Float16)v.x; h[1] = (_Float16)v.y;
    h[2] = (_Float16)v.z; h[3] = (_Float16)v.w;
    ((v4h*)y)[i] = h;
  }
}

// =====================================================================
// ROI align: one block per (cell, box), one thread per channel.
// roi layout: [m][c][cell]; writes f32 (for pooling) and f16 (GEMM A).
// =====================================================================
__global__ __launch_bounds__(256) void roi_align_kernel(
    const float* __restrict__ f2, const float* __restrict__ f3,
    const float* __restrict__ f4, const float* __restrict__ f5,
    const float* __restrict__ boxes,
    float* __restrict__ roi, _Float16* __restrict__ roih)
{
  const int cell = blockIdx.x;           // 0..48
  const int m    = blockIdx.y;           // 0..599
  const int c    = threadIdx.x;          // 0..255
  const int n    = m / NRP;

  const float bx0 = boxes[m * 4 + 0], by0 = boxes[m * 4 + 1];
  const float bx1 = boxes[m * 4 + 2], by1 = boxes[m * 4 + 3];
  const float area = (bx1 - bx0) * (by1 - by0);
  float lvf = floorf(4.0f + log2f(sqrtf(area) / 224.0f + 1e-8f));
  lvf = fminf(fmaxf(lvf, 2.0f), 5.0f);
  const int lv = (int)lvf - 2;

  const float* F; int H; float scale;
  if      (lv == 0) { F = f2; H = 200; scale = 0.25f;    }
  else if (lv == 1) { F = f3; H = 100; scale = 0.125f;   }
  else if (lv == 2) { F = f4; H = 50;  scale = 0.0625f;  }
  else              { F = f5; H = 25;  scale = 0.03125f; }
  const int W = H;

  const float x1b = bx0 * scale - 0.5f;
  const float y1b = by0 * scale - 0.5f;
  const float bw = (bx1 - bx0) * scale * (1.0f / 7.0f);
  const float bh = (by1 - by0) * scale * (1.0f / 7.0f);
  const int py = cell / 7, px = cell % 7;
  const float* Fc = F + (size_t)(n * DMOD + c) * H * W;

  float acc = 0.0f;
#pragma unroll
  for (int iy = 0; iy < 2; ++iy)
#pragma unroll
    for (int ix = 0; ix < 2; ++ix) {
      const float offy = ((float)(py * 2 + iy) + 0.5f) * 0.5f;
      const float offx = ((float)(px * 2 + ix) + 0.5f) * 0.5f;
      const float gy = y1b + offy * bh;
      const float gx = x1b + offx * bw;
      const bool valid = (gy >= -1.0f) && (gy <= (float)H) &&
                         (gx >= -1.0f) && (gx <= (float)W);
      const float y = fminf(fmaxf(gy, 0.0f), (float)(H - 1));
      const float x = fminf(fmaxf(gx, 0.0f), (float)(W - 1));
      const float y0f = floorf(y), x0f = floorf(x);
      const float ly = y - y0f, lx = x - x0f;
      const int y0 = (int)y0f, x0 = (int)x0f;
      const int y1 = min(y0 + 1, H - 1), x1 = min(x0 + 1, W - 1);
      const float v00 = Fc[y0 * W + x0], v01 = Fc[y0 * W + x1];
      const float v10 = Fc[y1 * W + x0], v11 = Fc[y1 * W + x1];
      float v = (1.0f - ly) * (1.0f - lx) * v00 + (1.0f - ly) * lx * v01 +
                ly * (1.0f - lx) * v10 + ly * lx * v11;
      acc += valid ? v : 0.0f;
    }
  const size_t idx = ((size_t)m * DMOD + c) * RR + cell;
  const float r = acc * 0.25f;
  roi[idx]  = r;
  roih[idx] = (_Float16)r;
}

// mean over the 49 cells -> pro[m][c] (f32 + f16)
__global__ __launch_bounds__(256) void pro_mean_kernel(
    const float* __restrict__ roi, float* __restrict__ pro,
    _Float16* __restrict__ proh)
{
  const int m = blockIdx.x, c = threadIdx.x;
  const float* r = roi + ((size_t)m * DMOD + c) * RR;
  float s = 0.0f;
#pragma unroll
  for (int k = 0; k < RR; ++k) s += r[k];
  const float v = s * (1.0f / 49.0f);
  pro[m * DMOD + c]  = v;
  proh[m * DMOD + c] = (_Float16)v;
}

// =====================================================================
// Attention: one wave per (proposal, head, image). qkv row = [q|k|v].
// Output f16 (consumed only as GEMM A operand).
// =====================================================================
__global__ __launch_bounds__(32) void attn_kernel(
    const float* __restrict__ qkv, _Float16* __restrict__ o)
{
  const int l = blockIdx.x, h = blockIdx.y, n = blockIdx.z;
  const int lane = threadIdx.x;
  __shared__ float sc[NRP];

  const float* q = qkv + (size_t)(n * NRP + l) * (3 * DMOD) + h * HD;
  for (int j = lane; j < NRP; j += 32) {
    const float* kj = qkv + (size_t)(n * NRP + j) * (3 * DMOD) + DMOD + h * HD;
    float a = 0.0f;
#pragma unroll
    for (int d = 0; d < HD; ++d) a += q[d] * kj[d];
    sc[j] = a * 0.17677669529663687f;  // 1/sqrt(32)
  }
  __syncthreads();

  float mx = -1e30f;
  for (int j = lane; j < NRP; j += 32) mx = fmaxf(mx, sc[j]);
  for (int t = 16; t; t >>= 1) mx = fmaxf(mx, __shfl_xor(mx, t, 32));
  float sum = 0.0f;
  for (int j = lane; j < NRP; j += 32) {
    const float e = __expf(sc[j] - mx);
    sc[j] = e;
    sum += e;
  }
  for (int t = 16; t; t >>= 1) sum += __shfl_xor(sum, t, 32);
  __syncthreads();

  const float inv = 1.0f / sum;
  const int d = lane;
  const float* vb = qkv + (size_t)n * NRP * (3 * DMOD) + 2 * DMOD + h * HD + d;
  float a = 0.0f;
  for (int j = 0; j < NRP; ++j) a += sc[j] * vb[(size_t)j * (3 * DMOD)];
  o[(size_t)(n * NRP + l) * DMOD + h * HD + d] = (_Float16)(a * inv);
}

// =====================================================================
// LayerNorm over last dim (dim <= 256), optional residual add + relu.
// Emits f32 (outF) and/or f16 (outH). One 256-thread block per row.
// =====================================================================
__global__ __launch_bounds__(256) void ln_kernel(
    const float* __restrict__ X, const float* __restrict__ res,
    const float* __restrict__ g, const float* __restrict__ b,
    float* __restrict__ outF, _Float16* __restrict__ outH,
    int dim, int relu)
{
  __shared__ float red[256];
  const int tid = threadIdx.x;
  const size_t base = (size_t)blockIdx.x * dim;
  float v = 0.0f;
  if (tid < dim) {
    v = X[base + tid];
    if (res) v += res[base + tid];
  }
  red[tid] = (tid < dim) ? v : 0.0f;
  __syncthreads();
  for (int s = 128; s; s >>= 1) { if (tid < s) red[tid] += red[tid + s]; __syncthreads(); }
  const float mean = red[0] / (float)dim;
  __syncthreads();
  const float d = (tid < dim) ? (v - mean) : 0.0f;
  red[tid] = d * d;
  __syncthreads();
  for (int s = 128; s; s >>= 1) { if (tid < s) red[tid] += red[tid + s]; __syncthreads(); }
  const float r = rsqrtf(red[0] / (float)dim + 1e-5f);
  if (tid < dim) {
    float ov = (v - mean) * r * g[tid] + b[tid];
    if (relu) ov = fmaxf(ov, 0.0f);
    if (outF) outF[base + tid] = ov;
    if (outH) outH[base + tid] = (_Float16)ov;
  }
}

__global__ __launch_bounds__(256) void silu_kernel(
    const float* __restrict__ x, _Float16* __restrict__ y, int n)
{
  const int i = blockIdx.x * 256 + threadIdx.x;
  if (i < n) { const float v = x[i]; y[i] = (_Float16)(v / (1.0f + __expf(-v))); }
}

// fc = obj * (scale + 1) + shift;  ss row = [scale(256) | shift(256)]
__global__ __launch_bounds__(256) void modulate_kernel(
    const float* __restrict__ obj, const float* __restrict__ ss,
    _Float16* __restrict__ fc)
{
  const int idx = blockIdx.x * 256 + threadIdx.x;   // < MTOT*DMOD
  const int m = idx >> 8, d = idx & 255, n = m / NRP;
  fc[idx] = (_Float16)(obj[idx] * (ss[n * 512 + d] + 1.0f) + ss[n * 512 + 256 + d]);
}

__global__ __launch_bounds__(64) void apply_deltas_kernel(
    const float* __restrict__ deltas, const float* __restrict__ boxes,
    float* __restrict__ pred)
{
  const int m = blockIdx.x * 64 + threadIdx.x;
  if (m >= MTOT) return;
  const float b0 = boxes[m * 4], b1 = boxes[m * 4 + 1];
  const float b2 = boxes[m * 4 + 2], b3 = boxes[m * 4 + 3];
  const float w = b2 - b0, h = b3 - b1;
  const float cx = b0 + 0.5f * w, cy = b1 + 0.5f * h;
  const float dx = deltas[m * 4] * 0.5f, dy = deltas[m * 4 + 1] * 0.5f;
  const float dw = fminf(deltas[m * 4 + 2], SCALE_CLAMP_F);
  const float dh = fminf(deltas[m * 4 + 3], SCALE_CLAMP_F);
  const float pcx = dx * w + cx, pcy = dy * h + cy;
  const float pw = __expf(dw) * w, ph = __expf(dh) * h;
  pred[m * 4 + 0] = pcx - 0.5f * pw;
  pred[m * 4 + 1] = pcy - 0.5f * ph;
  pred[m * 4 + 2] = pcx + 0.5f * pw;
  pred[m * 4 + 3] = pcy + 0.5f * ph;
}

// ---------------- host-side helpers ----------------
static inline void launch_gemm(hipStream_t s, int aT, int bT,
    const _Float16* A, long long sA, int lda,
    const _Float16* B, long long sB, int ldb,
    const float* bias, float* C, _Float16* Ch, long long sC, int ldc,
    int M, int N, int K, int nb, int relu)
{
  const int tiles = ((M + 15) / 16) * ((N + 63) / 64);
  dim3 grid((unsigned)((tiles + 7) / 8), (unsigned)nb, 1);
  if (aT == 0 && bT == 1)
    gemm_wmma_kernel<0, 1><<<grid, 256, 0, s>>>(A, sA, lda, B, sB, ldb, bias, C, Ch, sC, ldc, M, N, K, relu);
  else if (aT == 1 && bT == 0)
    gemm_wmma_kernel<1, 0><<<grid, 256, 0, s>>>(A, sA, lda, B, sB, ldb, bias, C, Ch, sC, ldc, M, N, K, relu);
  else
    gemm_wmma_kernel<0, 0><<<grid, 256, 0, s>>>(A, sA, lda, B, sB, ldb, bias, C, Ch, sC, ldc, M, N, K, relu);
}

static inline void cvt(hipStream_t s, const float* x, _Float16* y, long long n)
{
  const long long n4 = n / 4;
  cvt_f16_kernel<<<(unsigned)((n4 + 255) / 256), 256, 0, s>>>(x, y, (int)n4);
}

extern "C" void kernel_launch(void* const* d_in, const int* in_sizes, int n_in,
                              void* d_out, int out_size, void* d_ws, size_t ws_size,
                              hipStream_t stream)
{
  (void)in_sizes; (void)n_in; (void)out_size; (void)ws_size;
  // ------- inputs (setup_inputs dict order, params flattened) -------
  const float* F2i     = (const float*)d_in[0];
  const float* F3i     = (const float*)d_in[1];
  const float* F4i     = (const float*)d_in[2];
  const float* F5i     = (const float*)d_in[3];
  const float* bboxes  = (const float*)d_in[4];
  const float* time_e  = (const float*)d_in[5];
  const float* attn_Wi = (const float*)d_in[6];
  const float* attn_bi = (const float*)d_in[7];
  const float* attn_Wo = (const float*)d_in[8];
  const float* attn_bo = (const float*)d_in[9];
  const float* n1g = (const float*)d_in[10]; const float* n1b = (const float*)d_in[11];
  const float* n2g = (const float*)d_in[12]; const float* n2b = (const float*)d_in[13];
  const float* n3g = (const float*)d_in[14]; const float* n3b = (const float*)d_in[15];
  const float* dyn_W = (const float*)d_in[16]; const float* dyn_b = (const float*)d_in[17];
  const float* iin1g = (const float*)d_in[18]; const float* iin1b = (const float*)d_in[19];
  const float* iin2g = (const float*)d_in[20]; const float* iin2b = (const float*)d_in[21];
  const float* iioW  = (const float*)d_in[22]; const float* iiob  = (const float*)d_in[23];
  const float* iin3g = (const float*)d_in[24]; const float* iin3b = (const float*)d_in[25];
  const float* lin1W = (const float*)d_in[26]; const float* lin1b = (const float*)d_in[27];
  const float* lin2W = (const float*)d_in[28]; const float* lin2b = (const float*)d_in[29];
  const float* timeW = (const float*)d_in[30]; const float* timeb = (const float*)d_in[31];
  const float* clsW0 = (const float*)d_in[32];
  const float* clsg0 = (const float*)d_in[33]; const float* clsb0 = (const float*)d_in[34];
  const float* regW[3] = {(const float*)d_in[35], (const float*)d_in[36], (const float*)d_in[37]};
  const float* regG[3] = {(const float*)d_in[38], (const float*)d_in[39], (const float*)d_in[40]};
  const float* regB[3] = {(const float*)d_in[41], (const float*)d_in[42], (const float*)d_in[43]};
  const float* clW = (const float*)d_in[44]; const float* clB = (const float*)d_in[45];
  const float* dlW = (const float*)d_in[46]; const float* dlB = (const float*)d_in[47];

  float* out = (float*)d_out;
  float* outLogits = out;                 // (2,300,80)
  float* outPred   = out + 48000;         // (2,300,4)
  float* outObj    = out + 50400;         // (1,600,256)

  // ------- workspace layout: f32 region, then f16 region -------
  float* ws = (float*)d_ws;
  size_t o = 0;
  float* ROI  = ws + o; o += (size_t)MTOT * DMOD * RR;    // also reused as F2B
  float* PRO  = ws + o; o += (size_t)MTOT * DMOD;
  float* QKV  = ws + o; o += (size_t)MTOT * 3 * DMOD;
  float* TMP  = ws + o; o += (size_t)MTOT * DMOD;         // oproj out / ffn out
  float* X1   = ws + o; o += (size_t)MTOT * DMOD;
  float* F1B  = ws + o; o += (size_t)MTOT * RR * DDYN;
  float* DCV  = ws + o; o += (size_t)MTOT * DMOD;
  float* OBJA = ws + o; o += (size_t)MTOT * DMOD;
  float* SS   = ws + o; o += (size_t)NIMG * 2 * DMOD;
  float* CF   = ws + o; o += (size_t)MTOT * DMOD;
  float* RFf  = ws + o; o += (size_t)MTOT * DMOD;
  float* DLT  = ws + o; o += (size_t)MTOT * 4;
  float* F2B  = ROI;    // einsum2 f32 output reuses ROI buffer

  _Float16* hb = (_Float16*)(ws + o);
  size_t ho = 0;
  _Float16* ROIh  = hb + ho; ho += (size_t)MTOT * DMOD * RR;
  _Float16* PROh  = hb + ho; ho += (size_t)MTOT * DMOD;
  _Float16* AOh   = hb + ho; ho += (size_t)MTOT * DMOD;
  _Float16* X1h   = hb + ho; ho += (size_t)MTOT * DMOD;
  _Float16* PARh  = hb + ho; ho += (size_t)MTOT * NPAR;
  _Float16* F1Bh  = hb + ho; ho += (size_t)MTOT * RR * DDYN;
  _Float16* F2Bh  = hb + ho; ho += (size_t)MTOT * RR * DMOD;
  _Float16* OBJAh = hb + ho; ho += (size_t)MTOT * DMOD;
  _Float16* HBh   = hb + ho; ho += (size_t)MTOT * DFF_;
  _Float16* STh   = hb + ho; ho += (size_t)NIMG * 4 * DMOD;
  _Float16* FCh   = hb + ho; ho += (size_t)MTOT * DMOD;
  _Float16* CFh   = hb + ho; ho += (size_t)MTOT * DMOD;
  _Float16* RFh0  = hb + ho; ho += (size_t)MTOT * DMOD;
  _Float16* RFh1  = hb + ho; ho += (size_t)MTOT * DMOD;
  _Float16* Wih   = hb + ho; ho += (size_t)3 * DMOD * DMOD;
  _Float16* Woh   = hb + ho; ho += (size_t)DMOD * DMOD;
  _Float16* dynWh = hb + ho; ho += (size_t)NPAR * DMOD;
  _Float16* iioWh = hb + ho; ho += (size_t)DMOD * RR * DMOD;
  _Float16* lin1h = hb + ho; ho += (size_t)DFF_ * DMOD;
  _Float16* lin2h = hb + ho; ho += (size_t)DMOD * DFF_;
  _Float16* timeWh= hb + ho; ho += (size_t)2 * DMOD * 4 * DMOD;
  _Float16* clsWh = hb + ho; ho += (size_t)DMOD * DMOD;
  _Float16* regWh[3];
  for (int i = 0; i < 3; ++i) { regWh[i] = hb + ho; ho += (size_t)DMOD * DMOD; }
  _Float16* clWh  = hb + ho; ho += (size_t)80 * DMOD;
  _Float16* dlWh  = hb + ho; ho += (size_t)4 * DMOD;

  // 0) one-time weight conversions to f16
  cvt(stream, attn_Wi, Wih, 3 * DMOD * DMOD);
  cvt(stream, attn_Wo, Woh, DMOD * DMOD);
  cvt(stream, dyn_W, dynWh, (long long)NPAR * DMOD);
  cvt(stream, iioW, iioWh, (long long)DMOD * RR * DMOD);
  cvt(stream, lin1W, lin1h, DFF_ * DMOD);
  cvt(stream, lin2W, lin2h, DMOD * DFF_);
  cvt(stream, timeW, timeWh, 2 * DMOD * 4 * DMOD);
  cvt(stream, clsW0, clsWh, DMOD * DMOD);
  for (int i = 0; i < 3; ++i) cvt(stream, regW[i], regWh[i], DMOD * DMOD);
  cvt(stream, clW, clWh, 80 * DMOD);
  cvt(stream, dlW, dlWh, 4 * DMOD);

  // 1) ROI align + proposal pooling
  roi_align_kernel<<<dim3(RR, MTOT), 256, 0, stream>>>(F2i, F3i, F4i, F5i, bboxes, ROI, ROIh);
  pro_mean_kernel<<<MTOT, 256, 0, stream>>>(ROI, PRO, PROh);

  // 2) self-attention: qkv, softmax-attend, out-proj, residual + LN1
  launch_gemm(stream, 0, 1, PROh, 0, DMOD, Wih, 0, DMOD, attn_bi,
              QKV, nullptr, 0, 3 * DMOD, MTOT, 3 * DMOD, DMOD, 1, 0);
  attn_kernel<<<dim3(NRP, NHEAD, NIMG), 32, 0, stream>>>(QKV, AOh);
  launch_gemm(stream, 0, 1, AOh, 0, DMOD, Woh, 0, DMOD, attn_bo,
              TMP, nullptr, 0, DMOD, MTOT, DMOD, DMOD, 1, 0);
  ln_kernel<<<MTOT, 256, 0, stream>>>(TMP, PRO, n1g, n1b, X1, X1h, DMOD, 0);

  // 3) dynamic conv: par = x1 @ dyn_W^T + dyn_b  (600 x 32768, f16 out)
  launch_gemm(stream, 0, 1, X1h, 0, DMOD, dynWh, 0, DMOD, dyn_b,
              nullptr, PARh, 0, NPAR, MTOT, NPAR, DMOD, 1, 0);
  // einsum1 (batched, per proposal): (49x256) @ p1(256x64); A = roi col-major
  launch_gemm(stream, 1, 0, ROIh, (long long)DMOD * RR, RR,
              PARh, NPAR, DDYN, nullptr,
              F1B, nullptr, (long long)RR * DDYN, DDYN, RR, DDYN, DMOD, MTOT, 0);
  ln_kernel<<<MTOT * RR, 256, 0, stream>>>(F1B, nullptr, iin1g, iin1b,
                                           nullptr, F1Bh, DDYN, 1);
  // einsum2 (batched): (49x64) @ p2(64x256)
  launch_gemm(stream, 0, 0, F1Bh, (long long)RR * DDYN, DDYN,
              PARh + DMOD * DDYN, NPAR, DMOD, nullptr,
              F2B, nullptr, (long long)RR * DMOD, DMOD, RR, DMOD, DDYN, MTOT, 0);
  ln_kernel<<<MTOT * RR, 256, 0, stream>>>(F2B, nullptr, iin2g, iin2b,
                                           nullptr, F2Bh, DMOD, 1);
  // ii_out: (600 x 12544) @ (12544 x 256)
  launch_gemm(stream, 0, 1, F2Bh, 0, RR * DMOD, iioWh, 0, RR * DMOD, iiob,
              DCV, nullptr, 0, DMOD, MTOT, DMOD, RR * DMOD, 1, 0);
  ln_kernel<<<MTOT, 256, 0, stream>>>(DCV, nullptr, iin3g, iin3b, DCV, nullptr, DMOD, 1);
  // obj = LN(x1 + dconv, norm2)
  ln_kernel<<<MTOT, 256, 0, stream>>>(DCV, X1, n2g, n2b, OBJA, OBJAh, DMOD, 0);

  // 4) FFN + residual + LN3 (writes obj output)
  launch_gemm(stream, 0, 1, OBJAh, 0, DMOD, lin1h, 0, DMOD, lin1b,
              nullptr, HBh, 0, DFF_, MTOT, DFF_, DMOD, 1, 1);
  launch_gemm(stream, 0, 1, HBh, 0, DFF_, lin2h, 0, DFF_, lin2b,
              TMP, nullptr, 0, DMOD, MTOT, DMOD, DFF_, 1, 0);
  ln_kernel<<<MTOT, 256, 0, stream>>>(TMP, OBJA, n3g, n3b, outObj, nullptr, DMOD, 0);

  // 5) time scale/shift modulation
  silu_kernel<<<(NIMG * 4 * DMOD + 255) / 256, 256, 0, stream>>>(time_e, STh, NIMG * 4 * DMOD);
  launch_gemm(stream, 0, 1, STh, 0, 4 * DMOD, timeWh, 0, 4 * DMOD, timeb,
              SS, nullptr, 0, 2 * DMOD, NIMG, 2 * DMOD, 4 * DMOD, 1, 0);
  modulate_kernel<<<MTOT, 256, 0, stream>>>(outObj, SS, FCh);

  // 6) cls head (1 layer) and reg head (3 layers)
  launch_gemm(stream, 0, 1, FCh, 0, DMOD, clsWh, 0, DMOD, nullptr,
              CF, nullptr, 0, DMOD, MTOT, DMOD, DMOD, 1, 0);
  ln_kernel<<<MTOT, 256, 0, stream>>>(CF, nullptr, clsg0, clsb0, nullptr, CFh, DMOD, 1);

  const _Float16* rh = FCh;
  _Float16* rbuf[2] = {RFh0, RFh1};
  for (int i = 0; i < 3; ++i) {
    launch_gemm(stream, 0, 1, rh, 0, DMOD, regWh[i], 0, DMOD, nullptr,
                RFf, nullptr, 0, DMOD, MTOT, DMOD, DMOD, 1, 0);
    ln_kernel<<<MTOT, 256, 0, stream>>>(RFf, nullptr, regG[i], regB[i],
                                        nullptr, rbuf[i & 1], DMOD, 1);
    rh = rbuf[i & 1];
  }

  // 7) logits / deltas / boxes
  launch_gemm(stream, 0, 1, CFh, 0, DMOD, clWh, 0, DMOD, clB,
              outLogits, nullptr, 0, 80, MTOT, 80, DMOD, 1, 0);
  launch_gemm(stream, 0, 1, rh, 0, DMOD, dlWh, 0, DMOD, dlB,
              DLT, nullptr, 0, 4, MTOT, 4, DMOD, 1, 0);
  apply_deltas_kernel<<<(MTOT + 63) / 64, 64, 0, stream>>>(DLT, bboxes, outPred);
}
